// HistogramLoss_5239860101659
// MI455X (gfx1250) — matrix operand order repaired and checked by reading, not compiled, not run on hardware
//
#include <hip/hip_runtime.h>

typedef __attribute__((ext_vector_type(16))) _Float16 v16h;
typedef __attribute__((ext_vector_type(8)))  float    v8f;

#define BINS   256
#define PIX    16384            // 128*128
#define NROWS  24               // C*N = 3*8
#define WIN    20               // +/- bin window: exp(-0.0763*20^2) ~ 5e-14
#define WNORM  0.15585062f      // DELTA/(SIGMA*sqrt(2*pi)) = (1/256)/(0.01*2.5066283)

// ---------------------------------------------------------------------------
// Kernel 1: fused windowed Gaussian histogram + normalization.
// 48 blocks = (tensor in {input,target}) x (24 rows = channel*8 + n).
// LDS histogram + ds_add_f32 atomics; reads only the 3 MB of real input
// instead of materializing the (N,256,16384) diff tensor.
// ---------------------------------------------------------------------------
__global__ void __launch_bounds__(256)
hist_kernel(const float* __restrict__ inp, const float* __restrict__ tgt,
            float* __restrict__ ws) {
    __shared__ float sh[BINS];
    __shared__ float red[BINS];
    const int tid = threadIdx.x;
    const int blk = blockIdx.x;          // 0..47
    const int t   = blk / NROWS;         // 0 = input, 1 = target
    const int r   = blk % NROWS;         // r = c*8 + n
    const int c   = r >> 3;
    const int n   = r & 7;
    const float* src = ((t == 0) ? inp : tgt) + ((size_t)n * 3 + c) * PIX;

    sh[tid] = 0.0f;
    __syncthreads();

    for (int i = tid; i < PIX; i += 256) {
        const float x  = src[i];
        // nearest bin center index of x: centers are (b+0.5)/256
        const int b0 = (int)floorf(x * 256.0f);      // nearest-ish; window covers slack
        int lo = b0 - WIN; if (lo < 0)   lo = 0;
        int hi = b0 + WIN; if (hi > 255) hi = 255;
        for (int b = lo; b <= hi; ++b) {
            const float diff = x - ((float)b + 0.5f) * (1.0f / 256.0f);
            // -0.5/sigma^2 = -5000
            const float w = __expf(-5000.0f * diff * diff) * WNORM;
            atomicAdd(&sh[b], w);                     // ds_add_f32
        }
    }
    __syncthreads();

    red[tid] = sh[tid];
    __syncthreads();
    for (int s = 128; s > 0; s >>= 1) {
        if (tid < s) red[tid] += red[tid + s];
        __syncthreads();
    }
    const float total = red[0];
    ws[(size_t)blk * BINS + tid] = sh[tid] / (total + 1e-10f);
}

// ---------------------------------------------------------------------------
// Kernel 2: EMD surrogate via WMMA.
//   d[row,bin] = h_x - h_y                  (24 rows, padded to 32, x 256 bins)
//   C = cumsum(d) = D * U, U = upper-triangular ones (256x256)
// Computed transposed: C^T = U^T * D^T so the constant triangular A fragment
// is synthesized from lane IDs in registers (no memory), and only D^T is
// loaded. K-blocks with all-zero A (kb*32 > bin tile max) are skipped.
// One block, 8 waves, 32 output tiles (16 bin-tiles x 2 row-tiles).
// loss = (1/2048) * sum over rows,bins of C^2   (per-channel mean, summed).
// ---------------------------------------------------------------------------
__global__ void __launch_bounds__(256)
emd_wmma_kernel(const float* __restrict__ ws, float* __restrict__ out) {
    const float* __restrict__ hx = ws;
    const float* __restrict__ hy = ws + (size_t)NROWS * BINS;

    const int tid    = threadIdx.x;
    const int lane   = tid & 31;
    const int wave   = tid >> 5;          // 0..7
    const int l2     = lane & 15;
    const int hiHalf = (lane >= 16) ? 1 : 0;

    float sq = 0.0f;

    for (int tile = wave * 4; tile < wave * 4 + 4; ++tile) {
        const int mt = tile >> 1;                 // bin tile 0..15   (M of C^T)
        const int nt = tile & 1;                  // row tile 0..1    (N of C^T)
        const int mglob = mt * 16 + l2;           // this lane's A-matrix row (bin)
        const int row   = nt * 16 + l2;           // this lane's B-matrix col (hist row)
        const int rowValid = (row < NROWS) ? 1 : 0;
        const int rowC  = rowValid ? row : 0;     // clamped, masked by vmask
        const float vmask = rowValid ? 1.0f : 0.0f;

        v8f acc = {};
        const int kbMax = (mt * 16 + 15) >> 5;    // last kb with any nonzero A

        for (int kb = 0; kb <= kbMax; ++kb) {
            // ---- A fragment: U^T tile, A[m][k] = (k_glob <= m_glob) ? 1 : 0
            // layout: lanes 0-15 hold K=0..15, lanes 16-31 K=16..31
            v16h a;
            const int kA = hiHalf ? 16 : 0;
            #pragma unroll
            for (int j = 0; j < 8; ++j) {
                const int kg = kb * 32 + kA + j;
                a[j] = (kg <= mglob) ? (_Float16)1.0f : (_Float16)0.0f;
            }
            #pragma unroll
            for (int j = 0; j < 8; ++j) {
                const int kg = kb * 32 + kA + 8 + j;
                a[8 + j] = (kg <= mglob) ? (_Float16)1.0f : (_Float16)0.0f;
            }

            // ---- B fragment: D^T tile, B[k][n] = d[row n][bin k]
            // layout: lanes 0-15 hold K=kb*32+0..15 (2 per VGPR), lanes 16-31 K+16
            const int kOff = kb * 32 + (hiHalf ? 16 : 0);
            const float* __restrict__ px = hx + (size_t)rowC * BINS + kOff;
            const float* __restrict__ py = hy + (size_t)rowC * BINS + kOff;
            v16h b;
            #pragma unroll
            for (int j = 0; j < 16; ++j)
                b[j] = (_Float16)((px[j] - py[j]) * vmask);

            acc = __builtin_amdgcn_wmma_f32_16x16x32_f16(
                      /*neg_a=*/false, a, /*neg_b=*/false, b,
                      /*c_mod=*/(short)0, acc, /*reuse_a=*/false, /*reuse_b=*/false);
        }

        #pragma unroll
        for (int j = 0; j < 8; ++j) sq += acc[j] * acc[j];
    }

    __shared__ float red[256];
    red[tid] = sq;
    __syncthreads();
    for (int s = 128; s > 0; s >>= 1) {
        if (tid < s) red[tid] += red[tid + s];
        __syncthreads();
    }
    if (tid == 0) out[0] = red[0] * (1.0f / 2048.0f);
}

// ---------------------------------------------------------------------------
extern "C" void kernel_launch(void* const* d_in, const int* in_sizes, int n_in,
                              void* d_out, int out_size, void* d_ws, size_t ws_size,
                              hipStream_t stream) {
    (void)in_sizes; (void)n_in; (void)out_size; (void)ws_size;
    const float* inp = (const float*)d_in[0];   // (8,3,128,128) fp32
    const float* tgt = (const float*)d_in[1];   // (8,3,128,128) fp32
    float* ws  = (float*)d_ws;                  // 48 * 256 floats = 48 KB
    float* out = (float*)d_out;                 // scalar loss

    hist_kernel<<<48, 256, 0, stream>>>(inp, tgt, ws);
    emd_wmma_kernel<<<1, 256, 0, stream>>>(ws, out);
}
